// SWAttention_5781025980549
// MI455X (gfx1250) — matrix operand order repaired and checked by reading, not compiled, not run on hardware
//
#include <hip/hip_runtime.h>
#include <hip/hip_bf16.h>

// ---------------------------------------------------------------------------
// Types for CDNA5 WMMA
// ---------------------------------------------------------------------------
typedef __bf16 bf16;
typedef __attribute__((ext_vector_type(16))) __bf16 v16bf;
typedef __attribute__((ext_vector_type(8)))  __bf16 v8bf;
typedef __attribute__((ext_vector_type(8)))  float  v8f;
typedef __attribute__((ext_vector_type(4)))  int    v4i;

#define C_B    2
#define C_L    4096
#define C_HID  2048
#define C_H    32
#define C_KVH  8
#define C_D    64
#define C_W    512
#define C_NB   8
#define C_SCALE 0.125f  // 1/sqrt(64)

// gfx1250 async global->LDS copies (ASYNCcnt path), with safe fallback
#if defined(__gfx1250__) && __has_builtin(__builtin_amdgcn_global_load_async_to_lds_b128) && __has_builtin(__builtin_amdgcn_s_wait_asynccnt)
#define USE_ASYNC 1
#else
#define USE_ASYNC 0
#endif

#if USE_ASYNC
#define ASYNC_CP16(gsrc, ldst)                                            \
  __builtin_amdgcn_global_load_async_to_lds_b128(                         \
      (v4i*)(void*)(gsrc), (v4i*)(void*)(ldst), 0, 0)
#endif

__device__ __forceinline__ v8f zero8() {
  v8f z;
#pragma unroll
  for (int i = 0; i < 8; ++i) z[i] = 0.0f;
  return z;
}

__device__ __forceinline__ v16bf pack16(v8bf lo, v8bf hi) {
  v16bf r;
#pragma unroll
  for (int i = 0; i < 8; ++i) { r[i] = lo[i]; r[i + 8] = hi[i]; }
  return r;
}

// A-fragment (16x32 bf16): lane half 0 holds K 0..7 / 16..23, half 1 holds 8..15 / 24..31
__device__ __forceinline__ v16bf ld_frag_a(const bf16* rowp, int lhalf) {
  v8bf lo = *(const v8bf*)(rowp + lhalf * 8);
  v8bf hi = *(const v8bf*)(rowp + 16 + lhalf * 8);
  return pack16(lo, hi);
}

// B-fragment (32x16 bf16): lane = column, half 0 holds K 0..7 / 16..23, half 1 holds 8..15 / 24..31
__device__ __forceinline__ v16bf ld_frag_b(const bf16* colp, int lhalf) {
  v8bf lo = *(const v8bf*)(colp + lhalf * 8);
  v8bf hi = *(const v8bf*)(colp + 16 + lhalf * 8);
  return pack16(lo, hi);
}

__device__ __forceinline__ v8f wmma_bf16(v16bf a, v16bf b, v8f c) {
  return __builtin_amdgcn_wmma_f32_16x16x32_bf16(false, a, false, b, (short)0, c,
                                                 false, false);
}

// ---------------------------------------------------------------------------
// fp32 -> bf16 conversion
// ---------------------------------------------------------------------------
__global__ __launch_bounds__(256) void f32_to_bf16(const float* __restrict__ in,
                                                   bf16* __restrict__ out, int n) {
  int i = blockIdx.x * 256 + threadIdx.x;
  if (i < n) out[i] = (bf16)in[i];
}

// ---------------------------------------------------------------------------
// GEMM: C[M,N] = A[M,K] * Bw[N,K]^T   (bf16 in, fp32 accum, bf16 or fp32 out)
// Tile 128x128, 256 threads (8 waves, 2x4), K-step 32.
// Async path: double-buffered LDS, GLOBAL_LOAD_ASYNC_TO_LDS_B128 prefetch of
// tile t+1 overlapped with WMMA on tile t; one s_wait_asynccnt+barrier per step.
// ---------------------------------------------------------------------------
__global__ __launch_bounds__(256) void gemm_bf16_nt(
    const bf16* __restrict__ A, const bf16* __restrict__ Bw,
    float* __restrict__ Cf, bf16* __restrict__ Cb,
    int M, int N, int K, int write_f32) {
#if USE_ASYNC
  __shared__ bf16 As[2][128][40];   // pitch 80 B (16B aligned)
  __shared__ bf16 Bs[2][128][40];
#else
  __shared__ bf16 As[1][128][40];
  __shared__ bf16 Bs[1][128][40];
#endif

  const int tid   = threadIdx.x;
  const int lane  = tid & 31;
  const int wave  = tid >> 5;
  const int l15   = lane & 15;
  const int lhalf = lane >> 4;
  const int mw    = wave >> 2;   // 0..1
  const int nw    = wave & 3;    // 0..3
  const int mBase = blockIdx.y * 128;
  const int nBase = blockIdx.x * 128;

  v8f acc[4][2];
#pragma unroll
  for (int mi = 0; mi < 4; ++mi)
#pragma unroll
    for (int ni = 0; ni < 2; ++ni) acc[mi][ni] = zero8();

  auto compute = [&](int bsel) {
    v16bf afrag[4], bfrag[2];
#pragma unroll
    for (int mi = 0; mi < 4; ++mi)
      afrag[mi] = ld_frag_a(&As[bsel][mw * 64 + mi * 16 + l15][0], lhalf);
#pragma unroll
    for (int ni = 0; ni < 2; ++ni)
      bfrag[ni] = ld_frag_b(&Bs[bsel][nw * 32 + ni * 16 + l15][0], lhalf);
#pragma unroll
    for (int mi = 0; mi < 4; ++mi)
#pragma unroll
      for (int ni = 0; ni < 2; ++ni)
        acc[mi][ni] = wmma_bf16(afrag[mi], bfrag[ni], acc[mi][ni]);
  };

#if USE_ASYNC
  auto stage = [&](int bsel, int kt) {
#pragma unroll
    for (int i = 0; i < 2; ++i) {          // 512 16-byte chunks per operand tile
      int chunk = tid + i * 256;
      int row = chunk >> 2;
      int col = (chunk & 3) * 8;
      ASYNC_CP16(&A[(size_t)(mBase + row) * K + kt + col], &As[bsel][row][col]);
      ASYNC_CP16(&Bw[(size_t)(nBase + row) * K + kt + col], &Bs[bsel][row][col]);
    }
  };
  stage(0, 0);
  int buf = 0;
  for (int kt = 0; kt < K; kt += 32) {
    __builtin_amdgcn_s_wait_asynccnt(0);   // my copies into buf are done
    __syncthreads();                       // everyone's copies done; prev reads done
    if (kt + 32 < K) stage(buf ^ 1, kt + 32);
    compute(buf);
    buf ^= 1;
  }
#else
  for (int kt = 0; kt < K; kt += 32) {
    __syncthreads();
    unsigned long long ra[4], rb[4];
#pragma unroll
    for (int i = 0; i < 4; ++i) {          // 1024 8-byte chunks per tile
      int chunk = tid + i * 256;
      int row = chunk >> 3;
      int col = (chunk & 7) * 4;
      ra[i] = *(const unsigned long long*)&A[(size_t)(mBase + row) * K + kt + col];
      rb[i] = *(const unsigned long long*)&Bw[(size_t)(nBase + row) * K + kt + col];
    }
#pragma unroll
    for (int i = 0; i < 4; ++i) {
      int chunk = tid + i * 256;
      int row = chunk >> 3;
      int col = (chunk & 7) * 4;
      *(unsigned long long*)&As[0][row][col] = ra[i];
      *(unsigned long long*)&Bs[0][row][col] = rb[i];
    }
    __syncthreads();
    compute(0);
  }
#endif

#pragma unroll
  for (int mi = 0; mi < 4; ++mi)
#pragma unroll
    for (int ni = 0; ni < 2; ++ni) {
      int col = nBase + nw * 32 + ni * 16 + l15;
#pragma unroll
      for (int r = 0; r < 8; ++r) {
        int row = mBase + mw * 64 + mi * 16 + r + 8 * lhalf;
        float v = acc[mi][ni][r];
        if (write_f32) Cf[(size_t)row * N + col] = v;
        else           Cb[(size_t)row * N + col] = (bf16)v;
      }
    }
}

// ---------------------------------------------------------------------------
// RoPE (in-place on bf16 (B,L,nh,64)); one thread handles one (pos, head, d2) pair
// ---------------------------------------------------------------------------
__global__ __launch_bounds__(256) void rope_kernel(bf16* __restrict__ x, int nheads,
                                                   int total) {
  int idx = blockIdx.x * 256 + threadIdx.x;
  if (idx >= total) return;
  int d2 = idx & 31;
  int t  = idx >> 5;
  int h  = t % nheads;
  int t2 = t / nheads;             // b*L + l
  int l  = t2 & (C_L - 1);
  size_t base = (size_t)t2 * (nheads * 64) + h * 64;
  float x1 = (float)x[base + d2];
  float x2 = (float)x[base + d2 + 32];
  // inv = theta^(-d2/32); log2(10000) = 13.2877123795
  float freq = exp2f(-(float)d2 * (13.287712379549449f / 32.0f));
  float ang = (float)l * freq;
  float s, c;
  __sincosf(ang, &s, &c);
  x[base + d2]      = (bf16)(x1 * c - x2 * s);
  x[base + d2 + 32] = (bf16)(x2 * c + x1 * s);
}

// ---------------------------------------------------------------------------
// Sliding-window attention with tau sink. Block = (b,h,keyblock n, q-chunk of 128).
// 8 waves x 16 q rows. Online softmax over 5 key chunks of 128.
// ---------------------------------------------------------------------------
__global__ __launch_bounds__(256) void attn_kernel(
    const bf16* __restrict__ Qb,   // (B,L,H*64) roped
    const bf16* __restrict__ Kb,   // (B,L,KVH*64) roped
    const bf16* __restrict__ Vb,   // (B,L,KVH*64)
    const float* __restrict__ bias,// (H, W)
    const float* __restrict__ tau, // (H)
    bf16* __restrict__ Ob)         // (B,L,H*64)
{
  __shared__ bf16 Vts[64][136];        // V chunk transposed: [d][key], pitch 272 B
  __shared__ bf16 Ps[8][16][136];      // per-wave P staging, pitch 272 B
  __shared__ float bias_s[C_W];

  const int tid   = threadIdx.x;
  const int lane  = tid & 31;
  const int wave  = tid >> 5;
  const int l15   = lane & 15;
  const int lhalf = lane >> 4;

  const int idx = blockIdx.x;
  const int qc  = idx & 3;             // q chunk within 512-row block
  const int n   = (idx >> 2) & 7;      // key block
  const int h   = (idx >> 5) & 31;
  const int b   = idx >> 10;
  const int kvh = h >> 2;              // GQA group of 4

  for (int i = tid; i < C_W; i += 256) bias_s[i] = bias[h * C_W + i];
  const float tauv = tau[h];

  // ---- Q fragments (held in registers for all key chunks) ----
  const int rBase = wave * 16;
  v16bf qfrag[2];
  {
    int qrow = n * C_W + qc * 128 + rBase + l15;
    const bf16* qp = &Qb[((size_t)(b * C_L + qrow)) * C_HID + h * C_D];
#pragma unroll
    for (int ks = 0; ks < 2; ++ks) qfrag[ks] = ld_frag_a(qp + ks * 32, lhalf);
  }

  // ---- online softmax state (rows r+8*lhalf of this wave's 16) ----
  float m_run[8], s_run[8];
#pragma unroll
  for (int r = 0; r < 8; ++r) { m_run[r] = tauv; s_run[r] = 1.0f; } // sink term
  v8f accO[4];
#pragma unroll
  for (int nf = 0; nf < 4; ++nf) accO[nf] = zero8();

  // ---- 5 key chunks of 128 cover window (q_loc, q_loc+512] for all rows ----
  for (int kc = qc; kc <= qc + 4; ++kc) {
    __syncthreads();
    // cooperative transposed V chunk staging: batch 16B loads, then scatter
    v8bf vreg[4];
#pragma unroll
    for (int i = 0; i < 4; ++i) {       // 1024 8-elem chunks
      int chunk = tid + i * 256;
      int key = chunk >> 3;
      int d0 = (chunk & 7) * 8;
      int ki = kc * 128 + key;
      int lk = (n - 1) * C_W + ki; if (lk < 0) lk = 0;
      vreg[i] = *(const v8bf*)&Vb[((size_t)(b * C_L + lk)) * (C_KVH * C_D) + kvh * C_D + d0];
    }
#pragma unroll
    for (int i = 0; i < 4; ++i) {
      int chunk = tid + i * 256;
      int key = chunk >> 3;
      int d0 = (chunk & 7) * 8;
#pragma unroll
      for (int j = 0; j < 8; ++j) Vts[d0 + j][key] = vreg[i][j];
    }
    __syncthreads();

    // ---- S = Q K^T for 8 N-tiles of 16 keys ----
    float sv[8][8];
#pragma unroll
    for (int j = 0; j < 8; ++j) {
      int ki_col = kc * 128 + j * 16 + l15;
      int lk = (n - 1) * C_W + ki_col;
      if (lk < 0) lk = 0;
      const bf16* kp = &Kb[((size_t)(b * C_L + lk)) * (C_KVH * C_D) + kvh * C_D];
      v8f accS = zero8();
#pragma unroll
      for (int ks = 0; ks < 2; ++ks)
        accS = wmma_bf16(qfrag[ks], ld_frag_b(kp + ks * 32, lhalf), accS);
#pragma unroll
      for (int r = 0; r < 8; ++r) {
        int q_loc = qc * 128 + rBase + r + 8 * lhalf;        // in [0,512)
        int rel = q_loc - ki_col + C_W;
        bool ok = (rel >= 0) && (rel < C_W) && ((n > 0) || (ki_col >= C_W));
        int relc = rel < 0 ? 0 : (rel > C_W - 1 ? C_W - 1 : rel);
        sv[j][r] = ok ? (accS[r] * C_SCALE + bias_s[relc]) : -1e30f;
      }
    }

    // ---- online softmax update (xor-reduce inside 16-lane half) ----
    float alpha[8];
#pragma unroll
    for (int r = 0; r < 8; ++r) {
      float mx = sv[0][r];
#pragma unroll
      for (int j = 1; j < 8; ++j) mx = fmaxf(mx, sv[j][r]);
#pragma unroll
      for (int off = 1; off < 16; off <<= 1) mx = fmaxf(mx, __shfl_xor(mx, off, 32));
      float mn = fmaxf(m_run[r], mx);
      alpha[r] = __expf(m_run[r] - mn);
      m_run[r] = mn;
      float lsum = 0.0f;
#pragma unroll
      for (int j = 0; j < 8; ++j) {
        float p = __expf(sv[j][r] - mn);
        sv[j][r] = p;
        lsum += p;
      }
#pragma unroll
      for (int off = 1; off < 16; off <<= 1) lsum += __shfl_xor(lsum, off, 32);
      s_run[r] = s_run[r] * alpha[r] + lsum;
    }

    // ---- P -> LDS (wave-private), re-enter WMMA A layout ----
#pragma unroll
    for (int j = 0; j < 8; ++j)
#pragma unroll
      for (int r = 0; r < 8; ++r)
        Ps[wave][r + 8 * lhalf][j * 16 + l15] = (bf16)sv[j][r];

#pragma unroll
    for (int nf = 0; nf < 4; ++nf)
#pragma unroll
      for (int r = 0; r < 8; ++r) accO[nf][r] *= alpha[r];

    // ---- O += P @ Vchunk ----
#pragma unroll
    for (int ks = 0; ks < 4; ++ks) {
      v16bf pa = ld_frag_a(&Ps[wave][l15][ks * 32], lhalf);
#pragma unroll
      for (int nf = 0; nf < 4; ++nf) {
        v16bf vb = ld_frag_b(&Vts[nf * 16 + l15][ks * 32], lhalf);
        accO[nf] = wmma_bf16(pa, vb, accO[nf]);
      }
    }
  }

  // ---- epilogue: divide by denom, write bf16 O ----
  const int l_q0 = n * C_W + qc * 128 + rBase;
#pragma unroll
  for (int nf = 0; nf < 4; ++nf) {
    int dcol = nf * 16 + l15;
#pragma unroll
    for (int r = 0; r < 8; ++r) {
      int row = l_q0 + r + 8 * lhalf;
      float ov = accO[nf][r] / s_run[r];
      Ob[((size_t)(b * C_L + row)) * C_HID + h * C_D + dcol] = (bf16)ov;
    }
  }
}

// ---------------------------------------------------------------------------
// Launch
// ---------------------------------------------------------------------------
extern "C" void kernel_launch(void* const* d_in, const int* in_sizes, int n_in,
                              void* d_out, int out_size, void* d_ws, size_t ws_size,
                              hipStream_t stream) {
  const float* hidden = (const float*)d_in[0];
  const float* Wq = (const float*)d_in[1];
  const float* Wk = (const float*)d_in[2];
  const float* Wv = (const float*)d_in[3];
  const float* Wo = (const float*)d_in[4];
  const float* bias = (const float*)d_in[5];
  const float* tau = (const float*)d_in[6];
  float* out = (float*)d_out;

  char* ws = (char*)d_ws;
  bf16* h_bf  = (bf16*)(ws);               // 33,554,432 B
  bf16* wq_bf = (bf16*)(ws + 33554432);    //  8,388,608
  bf16* wk_bf = (bf16*)(ws + 41943040);    //  2,097,152
  bf16* wv_bf = (bf16*)(ws + 44040192);    //  2,097,152
  bf16* wo_bf = (bf16*)(ws + 46137344);    //  8,388,608
  bf16* q_bf  = (bf16*)(ws + 54525952);    // 33,554,432
  bf16* k_bf  = (bf16*)(ws + 88080384);    //  8,388,608
  bf16* v_bf  = (bf16*)(ws + 96468992);    //  8,388,608
  bf16* o_bf  = (bf16*)(ws + 104857600);   // 33,554,432  (total ~132 MB)

  f32_to_bf16<<<16777216 / 256, 256, 0, stream>>>(hidden, h_bf, 16777216);
  f32_to_bf16<<<4194304 / 256, 256, 0, stream>>>(Wq, wq_bf, 4194304);
  f32_to_bf16<<<1048576 / 256, 256, 0, stream>>>(Wk, wk_bf, 1048576);
  f32_to_bf16<<<1048576 / 256, 256, 0, stream>>>(Wv, wv_bf, 1048576);
  f32_to_bf16<<<4194304 / 256, 256, 0, stream>>>(Wo, wo_bf, 4194304);

  // Q/K/V projections (M = B*L = 8192)
  gemm_bf16_nt<<<dim3(16, 64), 256, 0, stream>>>(h_bf, wq_bf, nullptr, q_bf,
                                                 8192, 2048, 2048, 0);
  gemm_bf16_nt<<<dim3(4, 64), 256, 0, stream>>>(h_bf, wk_bf, nullptr, k_bf,
                                                8192, 512, 2048, 0);
  gemm_bf16_nt<<<dim3(4, 64), 256, 0, stream>>>(h_bf, wv_bf, nullptr, v_bf,
                                                8192, 512, 2048, 0);

  rope_kernel<<<8388608 / 256, 256, 0, stream>>>(q_bf, 32, 8388608);
  rope_kernel<<<2097152 / 256, 256, 0, stream>>>(k_bf, 8, 2097152);

  attn_kernel<<<2048, 256, 0, stream>>>(q_bf, k_bf, v_bf, bias, tau, o_bf);

  // output projection, fp32 straight to d_out
  gemm_bf16_nt<<<dim3(16, 64), 256, 0, stream>>>(o_bf, wo_bf, out, nullptr,
                                                 8192, 2048, 2048, 1);
}
